// GridNet_22909355556865
// MI455X (gfx1250) — compile-verified, matrix operand order checked by reading, not run on hardware
//
#include <hip/hip_runtime.h>

// ---------------------------------------------------------------------------
// GridNet on MI455X (gfx1250, wave32)
//
// Stage 1 (dominant, 201 MB input, ~7.3 GFLOP, HBM floor ~8.6us @ 23.3 TB/s):
//   per-patch conv(3->32, 3x3, s2, p1) + ReLU + GAP + FC(32->13) + bg-mask
//   as implicit GEMM on v_wmma_f32_16x16x32_f16:
//     M = 32 out-channels (2 tiles), N = 16 output pixels, K = 27 pad-> 32.
//   Patch staged ONCE in LDS as f16 with a zeroed halo [3][130][132] so the
//   B-operand gather needs NO bounds checks (unconditional ds_load_u16).
//   K-padding lanes load garbage-but-finite data and are nullified by the
//   zero elements already present in the A (weight) fragments.
//   ReLU + GAP fused on the accumulator: the [32,64,64] map never exists.
//
// Stage 2 (corrector, 53 KB, ~23 MFLOP): one 1024-thread block, one thread
//   per pixel, 13 channels in registers, map in static LDS, train-mode BN
//   via ds_add_f32 reductions + barriers. Tap-outer loop keeps code small.
// ---------------------------------------------------------------------------

typedef __attribute__((ext_vector_type(16))) _Float16 v16h;
typedef __attribute__((ext_vector_type(8)))  float    v8f;

#define NCLS   13
#define STEM   32
#define NPATCH 1024
#define CIN    3
#define PH     128
#define PW     128

// Halo-padded LDS layout: [c][row][col], data at [c][iy+1][ix+4]
#define SROW    132                       // padded row stride (halfs)
#define SROWS   130                       // padded rows
#define CSTRIDE (SROWS * SROW)            // 17160 halfs per channel
#define SPATCH_HALFS (CIN * CSTRIDE)      // 51480 halfs = 102,960 B

// ===========================================================================
// Kernel 1: one 256-thread block per patch (8 waves, 32 N-tiles per wave)
// ===========================================================================
__global__ __launch_bounds__(256)
void patch_classify_kernel(const float* __restrict__ x,
                           const float* __restrict__ pc_w,
                           const float* __restrict__ pc_b,
                           const float* __restrict__ fc_w,
                           const float* __restrict__ fc_b,
                           float* __restrict__ gridbuf)
{
    extern __shared__ _Float16 sPatch[];          // halo-padded patch, 100.5 KB
    __shared__ float sSums[STEM];
    __shared__ float sWaveMax[8];
    __shared__ float sPatchMax;

    const int tid   = threadIdx.x;
    const int lane  = tid & 31;
    const int wid   = tid >> 5;
    const int patch = blockIdx.x;
    const int m     = lane & 15;   // row-in-tile (A) / column (B)
    const int h     = lane >> 4;   // lane half, selects K sub-range

    if (tid < STEM) sSums[tid] = 0.0f;

    // ---- Phase A0: zero the whole padded buffer (halo = zeros) ------------
    {
        uint32_t* zp = reinterpret_cast<uint32_t*>(sPatch);
        for (int i = tid; i < SPATCH_HALFS / 2; i += 256) zp[i] = 0u;
    }
    __syncthreads();

    // ---- Phase A1: patch -> LDS interior (f32 -> f16), fused max ----------
    const float4* xin = reinterpret_cast<const float4*>(
        x + (size_t)patch * (CIN * PH * PW));
    float lmax = -3.0e38f;
    for (int i = tid; i < (CIN * PH * PW) / 4; i += 256) {
        float4 v = xin[i];
        lmax = fmaxf(lmax, fmaxf(fmaxf(v.x, v.y), fmaxf(v.z, v.w)));
        int rowid = i >> 5;                 // 32 float4 per 128-pixel row
        int c     = rowid >> 7;
        int iy    = rowid & 127;
        int col4  = (i & 31) << 2;
        _Float16 hv[4] = { (_Float16)v.x, (_Float16)v.y,
                           (_Float16)v.z, (_Float16)v.w };
        // byte offset = 2*(c*17160 + (iy+1)*132 + 4 + col4)  -> 8B aligned
        *reinterpret_cast<uint2*>(&sPatch[c*CSTRIDE + (iy+1)*SROW + 4 + col4]) =
            *reinterpret_cast<const uint2*>(hv);
    }
    for (int o = 16; o > 0; o >>= 1)
        lmax = fmaxf(lmax, __shfl_xor(lmax, o, 32));
    if (lane == 0) sWaveMax[wid] = lmax;
    __syncthreads();
    if (tid == 0) {
        float mx = sWaveMax[0];
        for (int i = 1; i < 8; ++i) mx = fmaxf(mx, sWaveMax[i]);
        sPatchMax = mx;
    }

    // ---- Phase B: A fragments + tile-invariant gather offsets -------------
    // CDNA5 16-bit operand layout: element e of lane (m + 16h) holds
    //   K(e) = 16*(e/8) + 8*h + (e&7);  K -> (c,ky,kx) = (K/9,(K%9)/3,K%3)
    // K >= 27: A element forced to 0, so B may load anything finite ->
    // point it at channel-0 data (baseOff = safe interior base).
    v16h a0, a1;
    int baseOff[16];
    #pragma unroll
    for (int e = 0; e < 16; ++e) {
        int K  = 16 * (e >> 3) + 8 * h + (e & 7);
        int ok = (K < 27);
        int c = 0, r = 0, s = 0;
        if (ok) { c = K / 9; int rem = K - 9 * c; r = rem / 3; s = rem - 3 * r; }
        // addr = c*CSTRIDE + (2*oy + r)*SROW + (2*ox + s + 3)
        baseOff[e] = ok ? (c * CSTRIDE + r * SROW + s + 3) : 3;
        a0[e] = ok ? (_Float16)pc_w[m * 27 + K]        : (_Float16)0.0f;
        a1[e] = ok ? (_Float16)pc_w[(16 + m) * 27 + K] : (_Float16)0.0f;
    }
    float bia0[8], bia1[8];
    #pragma unroll
    for (int j = 0; j < 8; ++j) {
        bia0[j] = pc_b[8 * h + j];
        bia1[j] = pc_b[16 + 8 * h + j];
    }

    __syncthreads();   // sPatch ready

    // ---- Phase C: WMMA conv, fused bias+ReLU+GAP --------------------------
    float acc0[8] = {0,0,0,0,0,0,0,0};
    float acc1[8] = {0,0,0,0,0,0,0,0};
    #pragma unroll 2
    for (int t = 0; t < 32; ++t) {          // uniform trip count: EXEC all-1s
        int tile = wid * 32 + t;
        int p    = tile * 16 + m;           // all 16 lanes share oy
        int oy   = p >> 6, ox = p & 63;
        int rowbase = oy * (2 * SROW) + 2 * ox;
        v16h b;
        #pragma unroll
        for (int e = 0; e < 16; ++e)        // 16 unconditional ds_load_u16
            b[e] = sPatch[baseOff[e] + rowbase];
        v8f c0 = {};
        v8f d0 = __builtin_amdgcn_wmma_f32_16x16x32_f16(
            false, a0, false, b, (short)0, c0, false, false);
        v8f c1 = {};
        v8f d1 = __builtin_amdgcn_wmma_f32_16x16x32_f16(
            false, a1, false, b, (short)0, c1, false, false);
        #pragma unroll
        for (int j = 0; j < 8; ++j) {       // D layout: vgpr j -> ch (j + 8h)
            acc0[j] += fmaxf(d0[j] + bia0[j], 0.0f);
            acc1[j] += fmaxf(d1[j] + bia1[j], 0.0f);
        }
    }
    #pragma unroll
    for (int j = 0; j < 8; ++j) {
        atomicAdd(&sSums[8 * h + j],      acc0[j]);
        atomicAdd(&sSums[16 + 8 * h + j], acc1[j]);
    }
    __syncthreads();

    // ---- Phase D: GAP -> FC -> bg mask -> grid[nc, gy, gx] ----------------
    if (tid < NCLS) {
        float logit = fc_b[tid];
        #pragma unroll
        for (int ch = 0; ch < STEM; ++ch)
            logit += (sSums[ch] * (1.0f / 4096.0f)) * fc_w[tid * STEM + ch];
        if (sPatchMax == 0.0f) logit = 0.0f;
        gridbuf[tid * NPATCH + patch] = logit;
    }
}

// ===========================================================================
// Kernel 2: corrector — one 1024-thread block, one thread per grid pixel.
// Tap-outer / oc-inner keeps out[13] register-indexed while ic/ky/kx stay
// rolled (small static code; uniform weight addresses -> scalar loads).
// ===========================================================================
template<int KS, int PAD, bool BNRELU>
__device__ __forceinline__
void corr_layer(const float* __restrict__ w, const float* __restrict__ bias,
                const float* __restrict__ gamma, const float* __restrict__ beta,
                float* buf, float* red, float* mv, int tid,
                float* __restrict__ outg)
{
    const int y = tid >> 5, xc = tid & 31;
    if (tid < 2 * NCLS) red[tid] = 0.0f;
    __syncthreads();

    float out[NCLS];
    #pragma unroll
    for (int oc = 0; oc < NCLS; ++oc) out[oc] = bias[oc];

    #pragma clang loop unroll(disable)
    for (int ic = 0; ic < NCLS; ++ic) {
        #pragma clang loop unroll(disable)
        for (int ky = 0; ky < KS; ++ky) {
            int iy = y + ky - PAD;
            if (iy < 0 || iy >= 32) continue;
            #pragma clang loop unroll(disable)
            for (int kx = 0; kx < KS; ++kx) {
                int ix = xc + kx - PAD;
                if (ix < 0 || ix >= 32) continue;
                float v = buf[ic * 1024 + iy * 32 + ix];
                const float* wp = w + ic * KS * KS + ky * KS + kx;
                #pragma unroll
                for (int oc = 0; oc < NCLS; ++oc)
                    out[oc] += wp[oc * NCLS * KS * KS] * v;
            }
        }
    }

    if (BNRELU) {
        #pragma unroll
        for (int oc = 0; oc < NCLS; ++oc) {
            atomicAdd(&red[oc],        out[oc]);            // ds_add_f32
            atomicAdd(&red[NCLS + oc], out[oc] * out[oc]);
        }
        __syncthreads();                 // all conv reads of buf done here
        if (tid < NCLS) {
            float mean = red[tid] * (1.0f / 1024.0f);
            float var  = red[NCLS + tid] * (1.0f / 1024.0f) - mean * mean;
            mv[tid]        = mean;
            mv[NCLS + tid] = __frsqrt_rn(var + 1e-5f);
        }
        __syncthreads();
        #pragma unroll
        for (int oc = 0; oc < NCLS; ++oc) {
            float v = (out[oc] - mv[oc]) * mv[NCLS + oc] * gamma[oc] + beta[oc];
            buf[oc * 1024 + tid] = fmaxf(v, 0.0f);
        }
        __syncthreads();
    } else {
        #pragma unroll
        for (int oc = 0; oc < NCLS; ++oc)
            outg[oc * 1024 + tid] = out[oc];
    }
}

__global__ __launch_bounds__(1024)
void corrector_kernel(const float* __restrict__ gridbuf,
                      const float* cw0, const float* cb0,
                      const float* cw1, const float* cb1,
                      const float* cw2, const float* cb2,
                      const float* cw3, const float* cb3,
                      const float* g0, const float* be0,
                      const float* g1, const float* be1,
                      const float* g2, const float* be2,
                      float* __restrict__ out)
{
    __shared__ float buf[NCLS * 1024];    // 52 KB resident feature map
    __shared__ float red[2 * NCLS];
    __shared__ float mv[2 * NCLS];
    const int tid = threadIdx.x;

    #pragma unroll
    for (int c = 0; c < NCLS; ++c) buf[c * 1024 + tid] = gridbuf[c * 1024 + tid];
    __syncthreads();

    corr_layer<3, 1, true >(cw0, cb0, g0, be0, buf, red, mv, tid, nullptr);
    corr_layer<5, 2, true >(cw1, cb1, g1, be1, buf, red, mv, tid, nullptr);
    corr_layer<5, 2, true >(cw2, cb2, g2, be2, buf, red, mv, tid, nullptr);
    corr_layer<3, 1, false>(cw3, cb3, nullptr, nullptr, buf, red, mv, tid, out);
}

// ===========================================================================
extern "C" void kernel_launch(void* const* d_in, const int* in_sizes, int n_in,
                              void* d_out, int out_size, void* d_ws, size_t ws_size,
                              hipStream_t stream)
{
    const float* x    = (const float*)d_in[0];
    const float* pc_w = (const float*)d_in[1];
    const float* pc_b = (const float*)d_in[2];
    const float* fc_w = (const float*)d_in[3];
    const float* fc_b = (const float*)d_in[4];
    const float* cw0  = (const float*)d_in[5];
    const float* cb0  = (const float*)d_in[6];
    const float* cw1  = (const float*)d_in[7];
    const float* cb1  = (const float*)d_in[8];
    const float* cw2  = (const float*)d_in[9];
    const float* cb2  = (const float*)d_in[10];
    const float* cw3  = (const float*)d_in[11];
    const float* cb3  = (const float*)d_in[12];
    const float* g0   = (const float*)d_in[13];
    const float* be0  = (const float*)d_in[14];
    const float* g1   = (const float*)d_in[15];
    const float* be1  = (const float*)d_in[16];
    const float* g2   = (const float*)d_in[17];
    const float* be2  = (const float*)d_in[18];

    float* gridbuf = (float*)d_ws;   // [13, 32, 32] intermediate

    const size_t shmem = (size_t)SPATCH_HALFS * sizeof(_Float16);  // 100.5 KB
    patch_classify_kernel<<<NPATCH, 256, shmem, stream>>>(
        x, pc_w, pc_b, fc_w, fc_b, gridbuf);

    corrector_kernel<<<1, 1024, 0, stream>>>(
        gridbuf, cw0, cb0, cw1, cb1, cw2, cb2, cw3, cb3,
        g0, be0, g1, be1, g2, be2, (float*)d_out);
}
